// GatingNetwork_65214783422489
// MI455X (gfx1250) — matrix operand order repaired and checked by reading, not compiled, not run on hardware
//
#include <hip/hip_runtime.h>

#define TOKENS_   16384
#define DIM_      4096
#define EXPERTS_  64

typedef __attribute__((ext_vector_type(16))) __bf16   v16bf;
typedef __attribute__((ext_vector_type(8)))  float    v8f;
typedef __attribute__((ext_vector_type(8)))  unsigned v8u;
typedef __attribute__((ext_vector_type(4)))  unsigned v4u;

// Packed fp32 -> 2x bf16 (RNE) in one VALU op: v_cvt_pk_bf16_f32 dst, lo, hi
static __device__ __forceinline__ unsigned cvt_pk_bf16(float lo, float hi) {
    unsigned r;
    asm("v_cvt_pk_bf16_f32 %0, %1, %2" : "=v"(r) : "v"(lo), "v"(hi));
    return r;
}

// A-tile fragment (this code's contiguous k order, same as B): per lane, K = {off..off+15}
// contiguous in VGPRs 0-7, 2 bf16 per VGPR.
static __device__ __forceinline__ v16bf loadA_bf(const float* __restrict__ p) {
    float4 q0 = *(const float4*)(p);
    float4 q1 = *(const float4*)(p + 4);
    float4 q2 = *(const float4*)(p + 8);
    float4 q3 = *(const float4*)(p + 12);
    v8u t;
    t[0] = cvt_pk_bf16(q0.x, q0.y);
    t[1] = cvt_pk_bf16(q0.z, q0.w);
    t[2] = cvt_pk_bf16(q1.x, q1.y);
    t[3] = cvt_pk_bf16(q1.z, q1.w);
    t[4] = cvt_pk_bf16(q2.x, q2.y);
    t[5] = cvt_pk_bf16(q2.z, q2.w);
    t[6] = cvt_pk_bf16(q3.x, q3.y);
    t[7] = cvt_pk_bf16(q3.z, q3.w);
    return __builtin_bit_cast(v16bf, t);
}

// Prologue: W [64,4096] fp32 -> bf16 in d_ws. 8 elements/thread.
__global__ __launch_bounds__(256, 1)
void convert_w_kernel(const float* __restrict__ W, unsigned* __restrict__ Wb) {
    const int i = blockIdx.x * 256 + threadIdx.x;
    const float4* s = (const float4*)W;
    float4 a = s[2 * i];
    float4 b = s[2 * i + 1];
    v4u o;
    o[0] = cvt_pk_bf16(a.x, a.y);
    o[1] = cvt_pk_bf16(a.z, a.w);
    o[2] = cvt_pk_bf16(b.x, b.y);
    o[3] = cvt_pk_bf16(b.z, b.w);
    *(v4u*)(Wb + 4 * (size_t)i) = o;
}

__global__ __launch_bounds__(128, 1)
void moe_gate_wmma_kernel(const float* __restrict__ x,
                          const __bf16* __restrict__ Wb,
                          const float* __restrict__ bias,
                          float* __restrict__ topw,
                          float* __restrict__ topi,
                          float* __restrict__ wts)
{
    // 4 waves/block; per wave: 32 tokens x 64 experts. Expert dim padded to 65
    // floats so the per-token row walk is bank-conflict-free ((m+e) mod 64).
    __shared__ float lds[4][32][EXPERTS_ + 1];

    const int wave = threadIdx.x >> 5;
    const int lane = threadIdx.x & 31;
    const int half = lane >> 4;   // 0 = lanes 0-15, 1 = lanes 16-31
    const int l16  = lane & 15;

    const int m0 = blockIdx.x * 128 + wave * 32;   // first token of this wave

    const float* xrow0 = x + (size_t)(m0 + l16) * DIM_ + half * 16;
    const float* xrow1 = xrow0 + (size_t)16 * DIM_;

    // B (32x16 bf16, contiguous k order same as A): lane l16 owns expert column t*16+l16, a
    // contiguous K-run of 16 offset by half*16 -> one 32-byte v16bf load.
    const __bf16* w0 = Wb + (size_t)( 0 + l16) * DIM_ + half * 16;
    const __bf16* w1 = Wb + (size_t)(16 + l16) * DIM_ + half * 16;
    const __bf16* w2 = Wb + (size_t)(32 + l16) * DIM_ + half * 16;
    const __bf16* w3 = Wb + (size_t)(48 + l16) * DIM_ + half * 16;

    v8f a00 = {}, a01 = {}, a02 = {}, a03 = {};   // tokens m0..m0+15
    v8f a10 = {}, a11 = {}, a12 = {}, a13 = {};   // tokens m0+16..m0+31

    for (int kb = 0; kb < DIM_; kb += 32) {
        v16bf A0 = loadA_bf(xrow0 + kb);
        v16bf A1 = loadA_bf(xrow1 + kb);
        v16bf B0 = *(const v16bf*)(w0 + kb);
        v16bf B1 = *(const v16bf*)(w1 + kb);
        v16bf B2 = *(const v16bf*)(w2 + kb);
        v16bf B3 = *(const v16bf*)(w3 + kb);
        a00 = __builtin_amdgcn_wmma_f32_16x16x32_bf16(false, A0, false, B0, (short)0, a00, false, false);
        a10 = __builtin_amdgcn_wmma_f32_16x16x32_bf16(false, A1, false, B0, (short)0, a10, false, false);
        a01 = __builtin_amdgcn_wmma_f32_16x16x32_bf16(false, A0, false, B1, (short)0, a01, false, false);
        a11 = __builtin_amdgcn_wmma_f32_16x16x32_bf16(false, A1, false, B1, (short)0, a11, false, false);
        a02 = __builtin_amdgcn_wmma_f32_16x16x32_bf16(false, A0, false, B2, (short)0, a02, false, false);
        a12 = __builtin_amdgcn_wmma_f32_16x16x32_bf16(false, A1, false, B2, (short)0, a12, false, false);
        a03 = __builtin_amdgcn_wmma_f32_16x16x32_bf16(false, A0, false, B3, (short)0, a03, false, false);
        a13 = __builtin_amdgcn_wmma_f32_16x16x32_bf16(false, A1, false, B3, (short)0, a13, false, false);
    }

    // Bias + scatter logits to LDS per the C/D layout:
    // lane l, acc element v -> token (v + 8*half [+16]), expert (t*16 + l16)
    const float be0 = bias[ 0 + l16];
    const float be1 = bias[16 + l16];
    const float be2 = bias[32 + l16];
    const float be3 = bias[48 + l16];
    #pragma unroll
    for (int v = 0; v < 8; ++v) {
        const int m = v + 8 * half;
        lds[wave][m][ 0 + l16]      = a00[v] + be0;
        lds[wave][m][16 + l16]      = a01[v] + be1;
        lds[wave][m][32 + l16]      = a02[v] + be2;
        lds[wave][m][48 + l16]      = a03[v] + be3;
        lds[wave][m + 16][ 0 + l16] = a10[v] + be0;
        lds[wave][m + 16][16 + l16] = a11[v] + be1;
        lds[wave][m + 16][32 + l16] = a12[v] + be2;
        lds[wave][m + 16][48 + l16] = a13[v] + be3;
    }

    __syncthreads();

    // Softmax + top-2: each lane owns one token.
    {
        const int m   = lane;
        const int tok = m0 + m;
        float* row = &lds[wave][m][0];

        float mx = row[0];
        #pragma unroll
        for (int e = 1; e < EXPERTS_; ++e) mx = fmaxf(mx, row[e]);

        float sum = 0.f;
        #pragma unroll
        for (int e = 0; e < EXPERTS_; ++e) {
            float pe = __expf(row[e] - mx);
            row[e] = pe;
            sum += pe;
        }
        const float inv = 1.f / sum;

        float w1v = -1.f, w2v = -1.f;
        int   i1 = 0,     i2 = 0;
        float4* wout = (float4*)(wts + (size_t)tok * EXPERTS_);
        #pragma unroll
        for (int e4 = 0; e4 < EXPERTS_ / 4; ++e4) {
            float4 pv;
            pv.x = row[e4 * 4 + 0] * inv;
            pv.y = row[e4 * 4 + 1] * inv;
            pv.z = row[e4 * 4 + 2] * inv;
            pv.w = row[e4 * 4 + 3] * inv;
            wout[e4] = pv;
            const float p[4] = {pv.x, pv.y, pv.z, pv.w};
            #pragma unroll
            for (int j = 0; j < 4; ++j) {
                const int e = e4 * 4 + j;
                if (p[j] > w1v)      { w2v = w1v; i2 = i1; w1v = p[j]; i1 = e; }
                else if (p[j] > w2v) { w2v = p[j]; i2 = e; }
            }
        }

        topw[tok * 2 + 0] = w1v;
        topw[tok * 2 + 1] = w2v;
        topi[tok * 2 + 0] = (float)i1;   // indices as float (single float d_out)
        topi[tok * 2 + 1] = (float)i2;
    }
}

extern "C" void kernel_launch(void* const* d_in, const int* in_sizes, int n_in,
                              void* d_out, int out_size, void* d_ws, size_t ws_size,
                              hipStream_t stream) {
    (void)in_sizes; (void)n_in; (void)out_size; (void)ws_size;
    const float* x = (const float*)d_in[0];
    const float* W = (const float*)d_in[1];
    const float* b = (const float*)d_in[2];

    float* out  = (float*)d_out;
    float* topw = out;                           // [16384, 2]
    float* topi = out + (size_t)TOKENS_ * 2;     // [16384, 2] (indices as float)
    float* wts  = out + (size_t)TOKENS_ * 4;     // [16384, 64]

    // Prologue: W -> bf16 in workspace (512 KB; re-done every call, no caching)
    convert_w_kernel<<<dim3((EXPERTS_ * DIM_ / 8) / 256), dim3(256), 0, stream>>>(
        W, (unsigned*)d_ws);

    // Main: 4 waves/block, 32 tokens/wave -> 128 tokens/block -> 128 blocks
    moe_gate_wmma_kernel<<<dim3(TOKENS_ / 128), dim3(128), 0, stream>>>(
        x, (const __bf16*)d_ws, b, topw, topi, wts);
}
